// SelfAttention2d_5669356834676
// MI455X (gfx1250) — compile-verified, matrix operand order checked by reading, not run on hardware
//
#include <hip/hip_runtime.h>

typedef __attribute__((ext_vector_type(16))) __bf16 v16bf;
typedef __attribute__((ext_vector_type(8)))  float  v8f;
typedef __attribute__((ext_vector_type(4)))  unsigned int v4u;
typedef __attribute__((ext_vector_type(8)))  int    v8i;
typedef __attribute__((ext_vector_type(4)))  int    v4i;

#define NB     8
#define CCH    512
#define NHEAD  8
#define DHEAD  64
#define HW     1024
#define GROUPS 32

static constexpr float QK_SCALE = 0.3535533905932738f; // 64^-0.25

// ---- workspace layout (bytes) ----
#define SS_OFF   (0ull)                 // 8*1024 f32
#define MEAN_OFF (64ull*1024)
#define RSTD_OFF (80ull*1024)
#define WB_OFF   (96ull*1024)           // bf16: Wqkv(1536*512) + Wout(512*512)
#define XT_OFF   (4ull*1024*1024)       // Xt  [8][1024][512] bf16
#define Q_OFF    (12ull*1024*1024)      // Q   [8][8][1024][64] bf16
#define K_OFF    (20ull*1024*1024)      // K   same
#define VT_OFF   (28ull*1024*1024)      // Vt  [8][8][64][1024] bf16
#define YT_OFF   (36ull*1024*1024)      // Yt  [8][1024][512] bf16

__device__ __forceinline__ v8f wmma_bf16(v16bf a, v16bf b, v8f c) {
  return __builtin_amdgcn_wmma_f32_16x16x32_bf16(false, a, false, b, (short)0, c, false, false);
}

// Issue one 2-D TDM load: tile (rows x 512 bf16 elems), row stride 512 elems,
// global -> LDS.  D# built per CDNA5 ISA.
// This toolchain's builtin takes 6 args (extra int32x8 group before cpol).
__device__ __forceinline__ void tdm_load_panel(const __bf16* gsrc,
                                               unsigned int lds_byte_off,
                                               int rows) {
  const unsigned long long ga = (unsigned long long)(size_t)gsrc;
  const unsigned int ga_lo = (unsigned int)__builtin_amdgcn_readfirstlane((int)(ga & 0xFFFFFFFFull));
  const unsigned int ga_hi = (unsigned int)__builtin_amdgcn_readfirstlane((int)((ga >> 32) & 0x01FFFFFFull));
  const unsigned int lds   = (unsigned int)__builtin_amdgcn_readfirstlane((int)lds_byte_off);
  // group 0: count=1 | lds_addr | global_addr | type=2 ("image")
  v4u g0;
  g0.x = 1u;                       // count=1, is_restore=0, gather_mode=0
  g0.y = lds;                      // lds_addr [63:32]
  g0.z = ga_lo;                    // global_addr [95:64]
  g0.w = ga_hi | (2u << 30);       // global_addr[56:32] | type=2
  // group 1: data_size=2B, tensor 512 x rows, tile 512 x rows, stride 512
  const unsigned int td0 = 512u, til0 = 512u, str0 = 512u;
  const unsigned int td1 = (unsigned int)rows, til1 = (unsigned int)rows;
  v8i g1;
  g1[0] = (int)(1u << 16);                                   // data_size=1 (2 bytes)
  g1[1] = (int)((td0 & 0xFFFFu) << 16);                      // tensor_dim0 lo
  g1[2] = (int)(((td0 >> 16) & 0xFFFFu) | ((td1 & 0xFFFFu) << 16));
  g1[3] = (int)(((td1 >> 16) & 0xFFFFu) | ((til0 & 0xFFFFu) << 16)); // tile_dim0
  g1[4] = (int)(til1 & 0xFFFFu);                             // tile_dim1, tile_dim2=0
  g1[5] = (int)str0;                                         // tensor_dim0_stride lo32
  g1[6] = 0;                                                 // stride hi / dim1_stride
  g1[7] = 0;
  v4i gz4 = {0, 0, 0, 0};
  v8i gz8 = {0, 0, 0, 0, 0, 0, 0, 0};
  __builtin_amdgcn_tensor_load_to_lds(g0, g1, gz4, gz4, gz8, 0);
}

// ---------- 1) ss = cond @ W_map^T + b_map ----------
__global__ void __launch_bounds__(256)
map_kernel(const float* __restrict__ cond, const float* __restrict__ Wm,
           const float* __restrict__ bm, float* __restrict__ ss) {
  const int n = blockIdx.x;
  const float* crow = cond + (size_t)n * CCH;
  for (int j = threadIdx.x; j < 2 * CCH; j += 256) {
    const float* wrow = Wm + (size_t)j * CCH;
    float acc = bm[j];
    #pragma unroll 4
    for (int c = 0; c < CCH; ++c) acc += crow[c] * wrow[c];
    ss[(size_t)n * 2 * CCH + j] = acc;
  }
}

// ---------- 2) group-norm stats ----------
__global__ void __launch_bounds__(256)
gn_stats_kernel(const float* __restrict__ x, float* __restrict__ mean,
                float* __restrict__ rstd) {
  __shared__ float s1[256], s2[256];
  const int b = blockIdx.x;
  const float* base = x + (size_t)b * 16 * HW;
  float a = 0.f, q = 0.f;
  for (int i = threadIdx.x; i < 16 * HW; i += 256) {
    float v = base[i]; a += v; q += v * v;
  }
  s1[threadIdx.x] = a; s2[threadIdx.x] = q;
  __syncthreads();
  for (int st = 128; st > 0; st >>= 1) {
    if (threadIdx.x < st) { s1[threadIdx.x] += s1[threadIdx.x + st];
                            s2[threadIdx.x] += s2[threadIdx.x + st]; }
    __syncthreads();
  }
  if (threadIdx.x == 0) {
    float m = s1[0] * (1.f / 16384.f);
    float v = s2[0] * (1.f / 16384.f) - m * m;
    mean[b] = m;
    rstd[b] = rsqrtf(v + 1e-5f);
  }
}

// ---------- 3) normalize + ada affine -> Xt[n][s][c] bf16 ----------
__global__ void __launch_bounds__(256)
norm_kernel(const float* __restrict__ x, const float* __restrict__ mean,
            const float* __restrict__ rstd, const float* __restrict__ ss,
            __bf16* __restrict__ Xt) {
  const int idx = blockIdx.x * 256 + threadIdx.x;
  const int n = idx >> 19;
  const int c = (idx >> 10) & (CCH - 1);
  const int s = idx & (HW - 1);
  const int b = n * GROUPS + (c >> 4);
  float v = (x[idx] - mean[b]) * rstd[b];
  v = v * (ss[(size_t)n * 2 * CCH + c] + 1.f) + ss[(size_t)n * 2 * CCH + CCH + c];
  Xt[((size_t)n * HW + s) * CCH + c] = (__bf16)v;
}

// ---------- 4) weights fp32 -> bf16 ----------
__global__ void __launch_bounds__(256)
wconv_kernel(const float* __restrict__ Wqkv, const float* __restrict__ Wout,
             __bf16* __restrict__ Wb) {
  const int idx = blockIdx.x * 256 + threadIdx.x;
  if (idx < 1536 * CCH) Wb[idx] = (__bf16)Wqkv[idx];
  else                  Wb[idx] = (__bf16)Wout[idx - 1536 * CCH];
}

// ---------- 5) qkv GEMM, TDM-staged B panel in LDS ----------
// Block: 128(o) x 64(s) of one batch; 8 waves x (16o x 64s each).
// B panel (64 s-rows x 512 c) = 64 KB LDS, loaded once by the Tensor Data Mover.
__global__ void __launch_bounds__(256)
qkv_gemm_kernel(const __bf16* __restrict__ Wb, const float* __restrict__ bq,
                const __bf16* __restrict__ Xt,
                __bf16* __restrict__ Q, __bf16* __restrict__ K,
                __bf16* __restrict__ Vt) {
  extern __shared__ __bf16 Bs[];                    // 64 * 512 bf16
  const int lane = threadIdx.x & 31, wave = threadIdx.x >> 5;
  const int r = lane & 15, half = lane >> 4;
  const int bx = blockIdx.x;                        // 8 * 12 * 16 blocks
  const int n  = bx / 192;
  const int rem = bx % 192;
  const int o0 = (rem / 16) * 128 + wave * 16;
  const int s0 = (rem & 15) * 64;

  if (threadIdx.x == 0)
    tdm_load_panel(Xt + ((size_t)n * HW + s0) * CCH,
                   (unsigned int)(size_t)&Bs[0], 64);
  __builtin_amdgcn_s_wait_tensorcnt(0);
  __syncthreads();

  const __bf16* arow = Wb + (size_t)(o0 + r) * CCH;
  v8f acc0 = {}, acc1 = {}, acc2 = {}, acc3 = {};
  for (int kk = 0; kk < CCH; kk += 32) {
    v16bf a = *reinterpret_cast<const v16bf*>(arow + kk + half * 16);
    __builtin_prefetch((const void*)(arow + kk + 128), 0, 1);
    v16bf b0 = *reinterpret_cast<const v16bf*>(&Bs[(size_t)(0 * 16 + r) * CCH + kk + half * 16]);
    v16bf b1 = *reinterpret_cast<const v16bf*>(&Bs[(size_t)(1 * 16 + r) * CCH + kk + half * 16]);
    v16bf b2 = *reinterpret_cast<const v16bf*>(&Bs[(size_t)(2 * 16 + r) * CCH + kk + half * 16]);
    v16bf b3 = *reinterpret_cast<const v16bf*>(&Bs[(size_t)(3 * 16 + r) * CCH + kk + half * 16]);
    acc0 = wmma_bf16(a, b0, acc0);
    acc1 = wmma_bf16(a, b1, acc1);
    acc2 = wmma_bf16(a, b2, acc2);
    acc3 = wmma_bf16(a, b3, acc3);
  }
  v8f accs[4] = {acc0, acc1, acc2, acc3};
  #pragma unroll
  for (int st = 0; st < 4; ++st) {
    const int s = s0 + st * 16 + r;
    #pragma unroll
    for (int i = 0; i < 8; ++i) {
      const int oo = o0 + i + 8 * half;
      const float val = accs[st][i] + bq[oo];
      const int tsel = oo >> 9;                     // uniform per block row
      const int hh = (oo & 511) >> 6;
      const int dd = oo & 63;
      const size_t sd = (((size_t)n * NHEAD + hh) * HW + s) * DHEAD + dd;
      if (tsel == 0)      Q[sd] = (__bf16)(val * QK_SCALE);
      else if (tsel == 1) K[sd] = (__bf16)(val * QK_SCALE);
      else Vt[(((size_t)n * NHEAD + hh) * DHEAD + dd) * HW + s] = (__bf16)val;
    }
  }
}

// ---------- 6) attention ----------
// One WG per (n,h,64 query rows).  S = 64x1024 f32 in LDS (256KB) + 16KB partials.
__global__ void __launch_bounds__(256)
attn_kernel(const __bf16* __restrict__ Q, const __bf16* __restrict__ K,
            const __bf16* __restrict__ Vt, __bf16* __restrict__ Yt) {
  extern __shared__ float S[];                      // 64*1024 f32 + 4*16*64 f32
  float* Pb = S + 64 * HW;                          // split-K partial buffer
  const int lane = threadIdx.x & 31, wave = threadIdx.x >> 5;
  const int r = lane & 15, half = lane >> 4;
  const int nh = blockIdx.x >> 4;
  const int qb = blockIdx.x & 15;
  const int n = nh >> 3, h = nh & 7;
  const __bf16* qp = Q  + (size_t)nh * HW * DHEAD;
  const __bf16* kp = K  + (size_t)nh * HW * DHEAD;
  const __bf16* vp = Vt + (size_t)nh * DHEAD * HW;

  // ---- S = q k^T : jobs of 16(q) x 64(k); A fragment reused across 4 WMMAs ----
  for (int t = wave; t < 64; t += 8) {
    const int m = t >> 4, ktb = t & 15;
    const __bf16* arow = qp + (size_t)(qb * 64 + m * 16 + r) * DHEAD;
    v8f acc[4] = {{}, {}, {}, {}};
    #pragma unroll
    for (int kk = 0; kk < DHEAD; kk += 32) {
      v16bf a = *reinterpret_cast<const v16bf*>(arow + kk + half * 16);
      #pragma unroll
      for (int dt = 0; dt < 4; ++dt) {
        v16bf b = *reinterpret_cast<const v16bf*>(
            kp + (size_t)(ktb * 64 + dt * 16 + r) * DHEAD + kk + half * 16);
        acc[dt] = wmma_bf16(a, b, acc[dt]);
      }
    }
    #pragma unroll
    for (int dt = 0; dt < 4; ++dt)
      #pragma unroll
      for (int i = 0; i < 8; ++i)
        S[(size_t)(m * 16 + i + 8 * half) * HW + ktb * 64 + dt * 16 + r] = acc[dt][i];
  }
  __syncthreads();

  // ---- softmax: 4 threads/row; repack P as bf16 in the low half of each slot ----
  {
    const int row = threadIdx.x >> 2, seg = threadIdx.x & 3;
    float* Srow = S + (size_t)row * HW;
    const int j0 = seg * 256;
    float mx = -3.0e38f;
    for (int j = j0; j < j0 + 256; ++j) mx = fmaxf(mx, Srow[j]);
    mx = fmaxf(mx, __shfl_xor(mx, 1, 32));
    mx = fmaxf(mx, __shfl_xor(mx, 2, 32));
    float sum = 0.f;
    for (int j = j0; j < j0 + 256; ++j) {
      float e = __expf(Srow[j] - mx);
      Srow[j] = e; sum += e;
    }
    sum += __shfl_xor(sum, 1, 32);
    sum += __shfl_xor(sum, 2, 32);
    const float inv = 1.f / sum;
    for (int j = j0; j < j0 + 256; ++j) {
      float p = Srow[j] * inv;
      *reinterpret_cast<__bf16*>(Srow + j) = (__bf16)p;
    }
  }
  __syncthreads();

  // ---- O = P V : split-K across wave pairs, LDS partial reduction ----
  {
    const int m = wave & 3;
    const int kk0 = (wave >> 2) * 512;
    v8f acc[4] = {{}, {}, {}, {}};
    for (int kk = kk0; kk < kk0 + 512; kk += 32) {
      union { v16bf v; __bf16 e[16]; } au;
      const float* ap = S + (size_t)(m * 16 + r) * HW + kk + half * 16;
      #pragma unroll
      for (int j = 0; j < 16; ++j)
        au.e[j] = *reinterpret_cast<const __bf16*>(ap + j);     // ds gather, reused 4x
      #pragma unroll
      for (int dt = 0; dt < 4; ++dt) {
        v16bf b = *reinterpret_cast<const v16bf*>(
            vp + (size_t)(dt * 16 + r) * HW + kk + half * 16);
        acc[dt] = wmma_bf16(au.v, b, acc[dt]);
      }
    }
    if (wave >= 4) {
      #pragma unroll
      for (int dt = 0; dt < 4; ++dt)
        #pragma unroll
        for (int i = 0; i < 8; ++i)
          Pb[(size_t)(m * 16 + i + 8 * half) * DHEAD + dt * 16 + r] = acc[dt][i];
    }
    __syncthreads();
    if (wave < 4) {
      #pragma unroll
      for (int dt = 0; dt < 4; ++dt) {
        #pragma unroll
        for (int i = 0; i < 8; ++i) {
          const float o = acc[dt][i] +
              Pb[(size_t)(m * 16 + i + 8 * half) * DHEAD + dt * 16 + r];
          const int qg = qb * 64 + m * 16 + i + 8 * half;
          const int cc = h * DHEAD + dt * 16 + r;
          Yt[((size_t)n * HW + qg) * CCH + cc] = (__bf16)o;
        }
      }
    }
  }
}

// ---------- 7) out = input + y @ W_out^T + b_out (16x64 per wave) ----------
__global__ void __launch_bounds__(256)
out_gemm_kernel(const __bf16* __restrict__ Wb, const float* __restrict__ bo,
                const __bf16* __restrict__ Yt, const float* __restrict__ x,
                float* __restrict__ out) {
  const int lane = threadIdx.x & 31, wave = threadIdx.x >> 5;
  const int r = lane & 15, half = lane >> 4;
  const int job = blockIdx.x * 8 + wave;            // 8 * 32 * 16 jobs
  const int n = job / 512;
  const int rem = job % 512;
  const int o0 = (rem >> 4) << 4;
  const int s0 = (rem & 15) << 6;
  const __bf16* arow = Wb + (size_t)(o0 + r) * CCH;
  v8f acc[4] = {{}, {}, {}, {}};
  for (int kk = 0; kk < CCH; kk += 32) {
    v16bf a = *reinterpret_cast<const v16bf*>(arow + kk + half * 16);
    __builtin_prefetch((const void*)(arow + kk + 128), 0, 1);
    #pragma unroll
    for (int st = 0; st < 4; ++st) {
      v16bf b = *reinterpret_cast<const v16bf*>(
          Yt + ((size_t)n * HW + s0 + st * 16 + r) * CCH + kk + half * 16);
      acc[st] = wmma_bf16(a, b, acc[st]);
    }
  }
  #pragma unroll
  for (int st = 0; st < 4; ++st) {
    #pragma unroll
    for (int i = 0; i < 8; ++i) {
      const int o = o0 + i + 8 * half;
      const size_t idx = ((size_t)n * CCH + o) * HW + s0 + st * 16 + r;
      out[idx] = x[idx] + acc[st][i] + bo[o];
    }
  }
}

extern "C" void kernel_launch(void* const* d_in, const int* in_sizes, int n_in,
                              void* d_out, int out_size, void* d_ws, size_t ws_size,
                              hipStream_t stream) {
  const float* x    = (const float*)d_in[0];
  const float* cond = (const float*)d_in[1];
  const float* Wqkv = (const float*)d_in[2];
  const float* bqkv = (const float*)d_in[3];
  const float* Wout = (const float*)d_in[4];
  const float* bout = (const float*)d_in[5];
  const float* Wmap = (const float*)d_in[6];
  const float* bmap = (const float*)d_in[7];
  float* out = (float*)d_out;
  char*  ws  = (char*)d_ws;

  float*  ss   = (float*)(ws + SS_OFF);
  float*  mean = (float*)(ws + MEAN_OFF);
  float*  rstd = (float*)(ws + RSTD_OFF);
  __bf16* Wb   = (__bf16*)(ws + WB_OFF);
  __bf16* Xt   = (__bf16*)(ws + XT_OFF);
  __bf16* Qb   = (__bf16*)(ws + Q_OFF);
  __bf16* Kb   = (__bf16*)(ws + K_OFF);
  __bf16* Vt   = (__bf16*)(ws + VT_OFF);
  __bf16* Yt   = (__bf16*)(ws + YT_OFF);

  map_kernel     <<<NB,          256, 0, stream>>>(cond, Wmap, bmap, ss);
  gn_stats_kernel<<<NB * GROUPS, 256, 0, stream>>>(x, mean, rstd);
  norm_kernel    <<<16384,       256, 0, stream>>>(x, mean, rstd, ss, Xt);
  wconv_kernel   <<<4096,        256, 0, stream>>>(Wqkv, Wout, Wb);
  qkv_gemm_kernel<<<1536,        256, 64 * CCH * sizeof(__bf16), stream>>>(
      Wb, bqkv, Xt, Qb, Kb, Vt);
  attn_kernel    <<<1024,        256,
      (64 * HW + 4 * 16 * DHEAD) * sizeof(float), stream>>>(Qb, Kb, Vt, Yt);
  out_gemm_kernel<<<512,         256, 0, stream>>>(
      Wb + 1536 * CCH, bout, Yt, x, out);
}
